// Encoder_7662221656405
// MI455X (gfx1250) — matrix operand order repaired and checked by reading, not compiled, not run on hardware
//
#include <hip/hip_runtime.h>
#include <hip/hip_bf16.h>
#include <math.h>

// ---------------- problem constants ----------------
#define BATCH    2000
#define NL       10
#define ROWS     (BATCH * NL)          // 20000
#define N_AID    200000
#define EMB      500
#define TDIM     50
#define HPAD     842
#define NEG      60000
#define NEG_TOPK 1000
#define TEMP     0.05f
#define INP_DIM  563                   // EMB + 3 + TDIM + NL
#define K1PAD    576                   // 18*32
#define EPAD     512                   // 16*32
#define HIDPAD   256                   // 250 -> 256
#define NBINS    2048
#define CHUNK    250                   // neg-GEMM row chunk (8 chunks)

typedef __attribute__((ext_vector_type(16))) _Float16 v16h;
typedef __attribute__((ext_vector_type(8)))  _Float16 v8h;
typedef __attribute__((ext_vector_type(8)))  float    v8f;

// ---------------- helpers ----------------
__device__ __forceinline__ float gelu_exact(float x) {
    return 0.5f * x * (1.0f + erff(x * 0.70710678118654752f));
}

__device__ __forceinline__ float block_reduce_sum(float v, float* red) {
    int tid = threadIdx.x;
    red[tid] = v; __syncthreads();
    for (int s = blockDim.x >> 1; s > 0; s >>= 1) {
        if (tid < s) red[tid] += red[tid + s];
        __syncthreads();
    }
    float r = red[0]; __syncthreads();
    return r;
}

__device__ __forceinline__ v16h wmma_step(v16h a, v16h b, v8f& acc) {
    acc = __builtin_amdgcn_wmma_f32_16x16x32_f16(
        false, a, false, b, (short)0, acc, false, false);
    return a;
}

// A fragment (16-bit A 16x32): lanes 0-15: K = kk+{0..15}; lanes 16-31: kk+{16..31}
__device__ __forceinline__ v16h load_a_frag(const _Float16* __restrict__ Arow, int kk, int hg) {
    const _Float16* pa = Arow + kk + hg * 16;
    v8h lo = *(const v8h*)pa;
    v8h hi = *(const v8h*)(pa + 8);
    v16h a;
    #pragma unroll
    for (int i = 0; i < 8; ++i) { a[i] = lo[i]; a[8 + i] = hi[i]; }
    return a;
}

// ---------------- 1) feature build ----------------
__global__ void build_features_kernel(const int* __restrict__ xy,
                                      const float* __restrict__ emb,
                                      const float* __restrict__ emb_h,
                                      float* __restrict__ Xf) {
    int r = blockIdx.x;                 // 0..ROWS-1
    int b = r / NL, l = r % NL;
    int aid = xy[b * 50 + l];
    int t   = xy[b * 50 + NL + l];
    int f   = xy[b * 50 + 2 * NL + l];
    float mask = (aid != (N_AID - 1)) ? 1.f : 0.f;
    float ang = (float)(t % 24) * (6.283185307179586f / 24.0f);
    int t1 = t + 1 < HPAD ? t + 1 : HPAD;
    int t2 = t + 2 < HPAD ? t + 2 : HPAD;
    for (int k = threadIdx.x; k < K1PAD; k += blockDim.x) {
        float v = 0.f;
        if (k < EMB)            v = emb[(size_t)aid * EMB + k];
        else if (k == EMB)      v = sinf(ang);
        else if (k == EMB + 1)  v = cosf(ang);
        else if (k == EMB + 2)  v = (float)f;
        else if (k < EMB + 3 + TDIM) {
            int d = k - (EMB + 3);
            v = emb_h[(size_t)t * TDIM + d] + emb_h[(size_t)t1 * TDIM + d]
              + emb_h[(size_t)t2 * TDIM + d];
        } else if (k < INP_DIM) {
            v = ((k - (EMB + 3 + TDIM)) == l) ? 1.f : 0.f;
        }
        Xf[(size_t)r * K1PAD + k] = v * mask;
    }
}

// ---------------- 2) batchnorm ----------------
__global__ void bn_stats_kernel(const float* __restrict__ Xf,
                                float* __restrict__ mean,
                                float* __restrict__ var) {
    __shared__ float red[256];
    int f = blockIdx.x;                 // 0..INP_DIM-1
    float s = 0.f, s2 = 0.f;
    for (int r = threadIdx.x; r < ROWS; r += blockDim.x) {
        float v = Xf[(size_t)r * K1PAD + f];
        s += v; s2 += v * v;
    }
    s  = block_reduce_sum(s, red);
    s2 = block_reduce_sum(s2, red);
    if (threadIdx.x == 0) {
        float mu = s / (float)ROWS;
        mean[f] = mu;
        var[f]  = s2 / (float)ROWS - mu * mu;
    }
}

__global__ void bn_apply_kernel(const float* __restrict__ Xf,
                                const float* __restrict__ mean,
                                const float* __restrict__ var,
                                const float* __restrict__ g,
                                const float* __restrict__ bb,
                                _Float16* __restrict__ Xh) {
    int r = blockIdx.x;
    for (int k = threadIdx.x; k < K1PAD; k += blockDim.x) {
        float y = 0.f;
        if (k < INP_DIM) {
            float v = Xf[(size_t)r * K1PAD + k];
            y = (v - mean[k]) * rsqrtf(var[k] + 1e-5f) * g[k] + bb[k];
        }
        Xh[(size_t)r * K1PAD + k] = (_Float16)y;
    }
}

// ---------------- weight transpose f32[K x N] -> f16 N-major [Npad x Kpad] ----------------
__global__ void convert_wt_kernel(const float* __restrict__ W,
                                  _Float16* __restrict__ Wt,
                                  int K, int N, int Kpad, int Npad) {
    int idx = blockIdx.x * blockDim.x + threadIdx.x;
    if (idx >= Kpad * Npad) return;
    int n = idx / Kpad, k = idx % Kpad;
    Wt[idx] = (k < K && n < N) ? (_Float16)W[(size_t)k * N + n] : (_Float16)0.f;
}

// ---------------- generic WMMA f16 GEMM ----------------
// C[M x N] = act(A[M x Kpad] * Wt^T + bias) (+ residual); Wt stored N-major [n*Kpad + k].
// 256 threads = 8 waves laid out 4(m) x 2(n); each wave owns a 32x64 output strip
// (2 m-tiles x 4 n-tiles = 8 WMMAs per k-step). Fast path is branch-free and
// software-pipelined (next k-step fragments in flight during current WMMAs).
__global__ void gemm_wmma_f16(const _Float16* __restrict__ A,
                              const _Float16* __restrict__ Wt,
                              const float* __restrict__ bias,
                              const float* __restrict__ residual,
                              float* __restrict__ Cf,
                              _Float16* __restrict__ Ch,
                              int M, int N, int Nreal, int Kpad, int ldC, int act) {
    const int lane = threadIdx.x & 31;
    const int wave = threadIdx.x >> 5;
    const int m0 = blockIdx.x * 128 + (wave & 3) * 32;
    const int n0 = blockIdx.y * 128 + (wave >> 2) * 64;
    if (m0 >= M || n0 >= N) return;            // wave-uniform
    const int hg = lane >> 4;
    const int lm = lane & 15;

    int row0 = m0 + lm;      if (row0 >= M) row0 = M - 1;
    int row1 = m0 + 16 + lm; if (row1 >= M) row1 = M - 1;
    const _Float16* Ar0 = A + (size_t)row0 * Kpad;
    const _Float16* Ar1 = A + (size_t)row1 * Kpad;

    v8f acc[2][4] = {};

    if (n0 + 64 <= N) {
        // ---- fast path: all 4 n-tiles valid ----
        const _Float16* Bp[4];
        #pragma unroll
        for (int j = 0; j < 4; ++j)
            Bp[j] = Wt + (size_t)(n0 + j * 16 + lm) * Kpad + hg * 16;

        v16h a0 = load_a_frag(Ar0, 0, hg);
        v16h a1 = load_a_frag(Ar1, 0, hg);
        v16h b0 = *(const v16h*)(Bp[0]);
        v16h b1 = *(const v16h*)(Bp[1]);
        v16h b2 = *(const v16h*)(Bp[2]);
        v16h b3 = *(const v16h*)(Bp[3]);

        for (int kk = 32; kk < Kpad; kk += 32) {
            v16h a0n = load_a_frag(Ar0, kk, hg);
            v16h a1n = load_a_frag(Ar1, kk, hg);
            v16h b0n = *(const v16h*)(Bp[0] + kk);
            v16h b1n = *(const v16h*)(Bp[1] + kk);
            v16h b2n = *(const v16h*)(Bp[2] + kk);
            v16h b3n = *(const v16h*)(Bp[3] + kk);
            wmma_step(a0, b0, acc[0][0]); wmma_step(a1, b0, acc[1][0]);
            wmma_step(a0, b1, acc[0][1]); wmma_step(a1, b1, acc[1][1]);
            wmma_step(a0, b2, acc[0][2]); wmma_step(a1, b2, acc[1][2]);
            wmma_step(a0, b3, acc[0][3]); wmma_step(a1, b3, acc[1][3]);
            a0 = a0n; a1 = a1n; b0 = b0n; b1 = b1n; b2 = b2n; b3 = b3n;
        }
        wmma_step(a0, b0, acc[0][0]); wmma_step(a1, b0, acc[1][0]);
        wmma_step(a0, b1, acc[0][1]); wmma_step(a1, b1, acc[1][1]);
        wmma_step(a0, b2, acc[0][2]); wmma_step(a1, b2, acc[1][2]);
        wmma_step(a0, b3, acc[0][3]); wmma_step(a1, b3, acc[1][3]);
    } else {
        // ---- slow path: partial N block column (rare) ----
        for (int kk = 0; kk < Kpad; kk += 32) {
            v16h a0 = load_a_frag(Ar0, kk, hg);
            v16h a1 = load_a_frag(Ar1, kk, hg);
            #pragma unroll
            for (int j = 0; j < 4; ++j) {
                int nt = n0 + j * 16;
                if (nt + 16 > N) break;        // wave-uniform
                v16h bf = *(const v16h*)(Wt + (size_t)(nt + lm) * Kpad + kk + hg * 16);
                wmma_step(a0, bf, acc[0][j]);
                wmma_step(a1, bf, acc[1][j]);
            }
        }
    }

    // ---- epilogue: bias + activation + residual + store ----
    #pragma unroll
    for (int mi = 0; mi < 2; ++mi) {
        #pragma unroll
        for (int j = 0; j < 4; ++j) {
            int nt = n0 + j * 16;
            if (nt + 16 > N) break;
            int n = nt + lm;
            float bv = (bias != nullptr && n < Nreal) ? bias[n] : 0.f;
            #pragma unroll
            for (int rI = 0; rI < 8; ++rI) {
                int m = m0 + mi * 16 + rI + hg * 8;
                if (m < M) {
                    float c = acc[mi][j][rI] + bv;
                    if (act == 1) c = gelu_exact(c);
                    if (residual) c += residual[(size_t)m * ldC + n];
                    if (Cf) Cf[(size_t)m * ldC + n] = c;
                    if (Ch) Ch[(size_t)m * ldC + n] = (_Float16)c;
                }
            }
        }
    }
}

// ---------------- mask rows after MLP2 ----------------
__global__ void mask_kernel(float* __restrict__ H, const int* __restrict__ xy) {
    int r = blockIdx.x;
    int b = r / NL, l = r % NL;
    if (xy[b * 50 + l] == (N_AID - 1)) {
        for (int k = threadIdx.x; k < EPAD; k += blockDim.x)
            H[(size_t)r * EPAD + k] = 0.f;
    }
}

// ---------------- layernorm ----------------
__global__ void ln_kernel(const float* __restrict__ H,
                          const float* __restrict__ g,
                          const float* __restrict__ bb,
                          float* __restrict__ HNf,
                          _Float16* __restrict__ HNh) {
    __shared__ float red[128];
    int r = blockIdx.x;
    const float* row = H + (size_t)r * EPAD;
    float s = 0.f, s2 = 0.f;
    for (int k = threadIdx.x; k < EMB; k += blockDim.x) {
        float v = row[k]; s += v; s2 += v * v;
    }
    s  = block_reduce_sum(s, red);
    s2 = block_reduce_sum(s2, red);
    float mu  = s / (float)EMB;
    float inv = rsqrtf(s2 / (float)EMB - mu * mu + 1e-5f);
    for (int k = threadIdx.x; k < EPAD; k += blockDim.x) {
        float v = (k < EMB) ? (row[k] - mu) * inv * g[k] + bb[k] : 0.f;
        HNf[(size_t)r * EPAD + k] = v;
        HNh[(size_t)r * EPAD + k] = (_Float16)v;
    }
}

// ---------------- masked mean pooling over NL ----------------
__global__ void pool_q_kernel(const float* __restrict__ H,
                              const int* __restrict__ xy,
                              float* __restrict__ q) {
    int b = blockIdx.x;
    float cnt = 0.f;
    bool msk[NL];
    for (int l = 0; l < NL; ++l) {
        msk[l] = xy[b * 50 + l] != (N_AID - 1);
        cnt += msk[l] ? 1.f : 0.f;
    }
    for (int n = threadIdx.x; n < EPAD; n += blockDim.x) {
        float s = 0.f;
        if (n < EMB) {
            for (int l = 0; l < NL; ++l)
                if (msk[l]) s += H[(size_t)(b * NL + l) * EPAD + n];
            s /= cnt;
        }
        q[(size_t)b * EPAD + n] = (n < EMB) ? s : 0.f;
    }
}

// ---------------- per-(b,l) cosine: cpos + x0_em (f16) ----------------
__global__ void cpa_kernel(const int* __restrict__ xy,
                           const float* __restrict__ q,
                           const float* __restrict__ emb_lt,
                           const float* __restrict__ emb,
                           float* __restrict__ cpos,
                           _Float16* __restrict__ X0h) {
    __shared__ float red[512];
    __shared__ float s_cmin, s_csum, s_ltsum;
    int b = blockIdx.x, tid = threadIdx.x;
    if (tid == 0) { s_cmin = 1e30f; s_csum = 0.f; s_ltsum = 0.f; }
    __syncthreads();
    const float* qb = q + (size_t)b * EPAD;
    for (int l = 0; l < NL; ++l) {
        int y  = xy[b * 50 + 3 * NL + l];
        int lt = xy[b * 50 + 4 * NL + l];
        const float* er = emb_lt + (size_t)lt * EMB;
        const float* yr = emb + (size_t)y * EMB;
        float sxx = 0.f, syy = 0.f, sxy = 0.f;
        for (int k = tid; k < EMB; k += blockDim.x) {
            float xe = qb[k] + er[k];
            float ye = yr[k];
            sxx += xe * xe; syy += ye * ye; sxy += xe * ye;
        }
        sxx = block_reduce_sum(sxx, red);
        syy = block_reduce_sum(syy, red);
        sxy = block_reduce_sum(sxy, red);
        float c = sxy * rsqrtf(sxx) * rsqrtf(syy);
        if (l == 0) {
            float inv = rsqrtf(sxx);
            for (int k = tid; k < EPAD; k += blockDim.x) {
                float xe = (k < EMB) ? (qb[k] + er[k]) * inv : 0.f;
                X0h[(size_t)b * EPAD + k] = (_Float16)xe;
            }
        }
        if (tid == 0) {
            float v = c + ((y == (N_AID - 1)) ? 1.f : 0.f);
            s_cmin = fminf(s_cmin, v);
            s_csum += c * (float)lt;
            s_ltsum += (float)lt;
        }
        __syncthreads();
    }
    if (tid == 0) cpos[b] = 0.5f * (s_csum / s_ltsum + s_cmin);
}

// ---------------- normalize negatives -> f16 ----------------
__global__ void neg_norm_kernel(const int* __restrict__ neg_aids,
                                const float* __restrict__ emb,
                                _Float16* __restrict__ Negh) {
    __shared__ float red[128];
    int j = blockIdx.x;
    int a = neg_aids[j];
    const float* e = emb + (size_t)a * EMB;
    float s = 0.f;
    for (int k = threadIdx.x; k < EMB; k += blockDim.x) { float v = e[k]; s += v * v; }
    s = block_reduce_sum(s, red);
    float inv = rsqrtf(s);
    for (int k = threadIdx.x; k < EPAD; k += blockDim.x)
        Negh[(size_t)j * EPAD + k] = (_Float16)((k < EMB) ? e[k] * inv : 0.f);
}

// ---------------- per-row top-k + log-softmax ----------------
__global__ void topk_loss_kernel(const float* __restrict__ scores,
                                 const float* __restrict__ cpos,
                                 int rowStart, float* __restrict__ accum) {
    __shared__ unsigned hist[NBINS];
    __shared__ float red[256];
    __shared__ int   sh_t, sh_need;
    __shared__ float sh_maxneg;
    int tid = threadIdx.x;
    int r = rowStart + blockIdx.x;
    const float* row = scores + (size_t)blockIdx.x * NEG;
    for (int i = tid; i < NBINS; i += blockDim.x) hist[i] = 0u;
    __syncthreads();
    float lmax = -2.f;
    for (int j = tid; j < NEG; j += blockDim.x) {
        float v = row[j];
        lmax = fmaxf(lmax, v);
        int bb = (int)((v + 1.f) * (NBINS * 0.5f));
        bb = bb < 0 ? 0 : (bb > NBINS - 1 ? NBINS - 1 : bb);
        atomicAdd(&hist[bb], 1u);
    }
    red[tid] = lmax; __syncthreads();
    for (int s = 128; s > 0; s >>= 1) {
        if (tid < s) red[tid] = fmaxf(red[tid], red[tid + s]);
        __syncthreads();
    }
    if (tid == 0) {
        sh_maxneg = red[0];
        unsigned cum = 0; int t = 0; unsigned chi = 0;
        for (int bb = NBINS - 1; bb >= 0; --bb) {
            unsigned c = hist[bb];
            if (cum + c >= NEG_TOPK) { t = bb; chi = cum; break; }
            cum += c;
        }
        sh_t = t; sh_need = (int)(NEG_TOPK - chi);
    }
    __syncthreads();
    float cp = cpos[r];
    float m = fmaxf(sh_maxneg, cp);
    int t = sh_t;
    float shi = 0.f, stt = 0.f;
    for (int j = tid; j < NEG; j += blockDim.x) {
        float v = row[j];
        int bb = (int)((v + 1.f) * (NBINS * 0.5f));
        bb = bb < 0 ? 0 : (bb > NBINS - 1 ? NBINS - 1 : bb);
        float e = __expf((v - m) * (1.f / TEMP));
        if (bb > t) shi += e;
        else if (bb == t) stt += e;
    }
    shi = block_reduce_sum(shi, red);
    stt = block_reduce_sum(stt, red);
    if (tid == 0) {
        float ct = (float)hist[t];
        float S = __expf((cp - m) * (1.f / TEMP)) + shi + stt * ((float)sh_need / ct);
        float loss = logf(S) - (cp - m) * (1.f / TEMP);
        float acc = (cp >= sh_maxneg) ? 1.f : 0.f;
        atomicAdd(&accum[0], loss);
        atomicAdd(&accum[1], acc);
    }
}

__global__ void zero2_kernel(float* a) { if (threadIdx.x < 2) a[threadIdx.x] = 0.f; }

__global__ void finalize_kernel(const float* __restrict__ accum, float* __restrict__ out) {
    if (threadIdx.x == 0) {
        out[0] = accum[0] / (float)BATCH;
        out[1] = accum[1] / (float)BATCH;
    }
}

// ---------------- host launch ----------------
extern "C" void kernel_launch(void* const* d_in, const int* in_sizes, int n_in,
                              void* d_out, int out_size, void* d_ws, size_t ws_size,
                              hipStream_t stream) {
    (void)in_sizes; (void)n_in; (void)out_size; (void)ws_size;
    const int*   xy       = (const int*)d_in[0];
    const int*   neg_aids = (const int*)d_in[1];
    const float* emb      = (const float*)d_in[2];
    const float* emb_h    = (const float*)d_in[3];
    const float* emb_lt   = (const float*)d_in[4];
    const float* bn_g     = (const float*)d_in[5];
    const float* bn_b     = (const float*)d_in[6];
    const float* mlp_w1   = (const float*)d_in[7];
    const float* mlp_b1   = (const float*)d_in[8];
    const float* mlp_w2   = (const float*)d_in[9];
    const float* mlp_b2   = (const float*)d_in[10];
    float* out = (float*)d_out;

    // workspace carve-up (256B aligned); regionA/regionB are time-aliased
    char* w = (char*)d_ws;
    auto alloc = [&](size_t bytes) {
        char* p = w; w += (bytes + 255) & ~(size_t)255; return p;
    };
    _Float16* Xh     = (_Float16*)alloc((size_t)ROWS * K1PAD * 2);
    char*     regA   = alloc((size_t)CHUNK * NEG * 4);            // 60 MB: Xf(46)->Hf32(41)->scores(60)
    char*     regB   = alloc((size_t)NEG * EPAD * 2);             // 61.4 MB: HNf32(41)->Negh(61.4)
    _Float16* H1h    = (_Float16*)alloc((size_t)ROWS * EPAD * 2); // also reused as HNh
    _Float16* Th     = (_Float16*)alloc((size_t)ROWS * HIDPAD * 2);
    _Float16* W1t    = (_Float16*)alloc((size_t)EPAD * K1PAD * 2);
    _Float16* W2t    = (_Float16*)alloc((size_t)EPAD * EPAD * 2);
    _Float16* Bw1t[3], *Bw2t[3];
    for (int i = 0; i < 3; ++i) {
        Bw1t[i] = (_Float16*)alloc((size_t)HIDPAD * EPAD * 2);
        Bw2t[i] = (_Float16*)alloc((size_t)EPAD * HIDPAD * 2);
    }
    float* meanv = (float*)alloc(K1PAD * 4);
    float* varv  = (float*)alloc(K1PAD * 4);
    float* qbuf  = (float*)alloc((size_t)BATCH * EPAD * 4);
    _Float16* X0h = (_Float16*)alloc((size_t)BATCH * EPAD * 2);
    float* cposb = (float*)alloc(BATCH * 4);
    float* accum = (float*)alloc(256);

    float* Xf     = (float*)regA;       // [ROWS x K1PAD]
    float* Hf32   = (float*)regA;       // [ROWS x EPAD] (after Xf dead)
    float* scores = (float*)regA;       // [CHUNK x NEG] (after Hf32 dead)
    float* HNf    = (float*)regB;       // [ROWS x EPAD]
    _Float16* Negh = (_Float16*)regB;   // [NEG x EPAD] (after HNf dead)
    _Float16* HNh = H1h;                // reuse f16 activation buffer

    // 1) features (f32) -> 2) BN stats -> 3) BN apply (f16)
    build_features_kernel<<<ROWS, 128, 0, stream>>>(xy, emb, emb_h, Xf);
    bn_stats_kernel<<<INP_DIM, 256, 0, stream>>>(Xf, meanv, varv);
    bn_apply_kernel<<<ROWS, 128, 0, stream>>>(Xf, meanv, varv, bn_g, bn_b, Xh);

    // weight transposes to f16 N-major
    auto cvt = [&](const float* W, _Float16* Wt, int K, int N, int Kp, int Np) {
        int tot = Kp * Np;
        convert_wt_kernel<<<(tot + 255) / 256, 256, 0, stream>>>(W, Wt, K, N, Kp, Np);
    };
    cvt(mlp_w1, W1t, INP_DIM, EMB, K1PAD, EPAD);
    cvt(mlp_w2, W2t, EMB, EMB, EPAD, EPAD);
    for (int i = 0; i < 3; ++i) {
        cvt((const float*)d_in[13 + 6 * i], Bw1t[i], EMB, 250, EPAD, HIDPAD);
        cvt((const float*)d_in[15 + 6 * i], Bw2t[i], 250, EMB, HIDPAD, EPAD);
    }

    dim3 blk(256);
    auto grd = [](int M, int N) { return dim3((M + 127) / 128, (N + 127) / 128); };
    // MLP1: gelu -> f16
    gemm_wmma_f16<<<grd(ROWS, EPAD), blk, 0, stream>>>(
        Xh, W1t, mlp_b1, nullptr, nullptr, H1h, ROWS, EPAD, EMB, K1PAD, EPAD, 1);
    // MLP2 -> f32 h
    gemm_wmma_f16<<<grd(ROWS, EPAD), blk, 0, stream>>>(
        H1h, W2t, mlp_b2, nullptr, Hf32, nullptr, ROWS, EPAD, EMB, EPAD, EPAD, 0);
    mask_kernel<<<ROWS, 512, 0, stream>>>(Hf32, xy);

    // 3 residual blocks
    for (int i = 0; i < 3; ++i) {
        const float* ln_g = (const float*)d_in[11 + 6 * i];
        const float* ln_b = (const float*)d_in[12 + 6 * i];
        const float* b1   = (const float*)d_in[14 + 6 * i];
        const float* b2   = (const float*)d_in[16 + 6 * i];
        ln_kernel<<<ROWS, 128, 0, stream>>>(Hf32, ln_g, ln_b, HNf, HNh);
        gemm_wmma_f16<<<grd(ROWS, HIDPAD), blk, 0, stream>>>(
            HNh, Bw1t[i], b1, nullptr, nullptr, Th, ROWS, HIDPAD, 250, EPAD, HIDPAD, 1);
        gemm_wmma_f16<<<grd(ROWS, EPAD), blk, 0, stream>>>(
            Th, Bw2t[i], b2, HNf, Hf32, nullptr, ROWS, EPAD, EMB, HIDPAD, EPAD, 0);
    }

    // pooling, positives, x0 embedding
    pool_q_kernel<<<BATCH, 512, 0, stream>>>(Hf32, xy, qbuf);
    cpa_kernel<<<BATCH, 512, 0, stream>>>(xy, qbuf, emb_lt, emb, cposb, X0h);

    // negatives (regB now free of HNf)
    neg_norm_kernel<<<NEG, 128, 0, stream>>>(neg_aids, emb, Negh);

    zero2_kernel<<<1, 32, 0, stream>>>(accum);

    // chunked cneg GEMM (WMMA) + fused top-k/log-softmax (regA now free of Hf32)
    for (int c = 0; c < BATCH / CHUNK; ++c) {
        gemm_wmma_f16<<<grd(CHUNK, NEG), blk, 0, stream>>>(
            X0h + (size_t)c * CHUNK * EPAD, Negh, nullptr, nullptr,
            scores, nullptr, CHUNK, NEG, NEG, EPAD, NEG, 0);
        topk_loss_kernel<<<CHUNK, 256, 0, stream>>>(scores, cposb, c * CHUNK, accum);
    }
    finalize_kernel<<<1, 32, 0, stream>>>(accum, out);
}
